// GLABlock_23398981829041
// MI455X (gfx1250) — matrix operand order repaired and checked by reading, not compiled, not run on hardware
//
#include <hip/hip_runtime.h>
#include <hip/hip_bf16.h>

typedef __bf16 bf16_t;
typedef __attribute__((ext_vector_type(16))) __bf16 v16bf;
typedef __attribute__((ext_vector_type(8)))  __bf16 v8bf;
typedef __attribute__((ext_vector_type(8)))  float   v8f;

union Frag16 { v16bf v; v8bf h[2]; bf16_t e[16]; };
union Acc8   { v8f   v; float f[8]; };

#define D_DIM 1024
#define T_DIM 4096
#define NROW  16384L   // B*T

// ---------------------------------------------------------------- cvt f32->bf16
__global__ __launch_bounds__(256)
void cvt_bf16_kernel(const float* __restrict__ src, bf16_t* __restrict__ dst, long n) {
    long i = (long)blockIdx.x * blockDim.x + threadIdx.x;
    long stride = (long)gridDim.x * blockDim.x;
    for (; i < n; i += stride) dst[i] = (bf16_t)src[i];
}

// ---------------------------------------------------------------- dual LayerNorm
__global__ __launch_bounds__(256)
void ln_dual_kernel(const float* __restrict__ x,
                    const float* __restrict__ ln_g, const float* __restrict__ ln_b,
                    const float* __restrict__ gg,   const float* __restrict__ gb,
                    bf16_t* __restrict__ xnB, bf16_t* __restrict__ gxB) {
    const long row = blockIdx.x;                  // 16384 rows
    const float* xr = x + row * D_DIM;
    float s = 0.f, s2 = 0.f;
    for (int i = threadIdx.x; i < D_DIM; i += 256) { float v = xr[i]; s += v; s2 += v * v; }
    __shared__ float r1[256], r2[256];
    r1[threadIdx.x] = s; r2[threadIdx.x] = s2; __syncthreads();
    for (int off = 128; off > 0; off >>= 1) {
        if (threadIdx.x < off) { r1[threadIdx.x] += r1[threadIdx.x + off];
                                 r2[threadIdx.x] += r2[threadIdx.x + off]; }
        __syncthreads();
    }
    const float mean = r1[0] * (1.f / D_DIM);
    const float var  = r2[0] * (1.f / D_DIM) - mean * mean;
    const float inv  = rsqrtf(var + 1e-5f);
    for (int i = threadIdx.x; i < D_DIM; i += 256) {
        float v = (xr[i] - mean) * inv;
        xnB[row * D_DIM + i] = (bf16_t)(v * ln_g[i] + ln_b[i]);
        gxB[row * D_DIM + i] = (bf16_t)(v * gg[i]  + gb[i]);
    }
}

// ---------------------------------------------------------------- main bf16 GEMM
// A: [16384][K] bf16 row-major; W: [N][K] bf16 row-major (torch Linear weight)
// block tile 128x128, 8 waves (4 M x 2 N), wave tile 32x64 = 2x4 WMMA tiles
__global__ __launch_bounds__(256)
void gemm_bf16_kernel(const bf16_t* __restrict__ A, const bf16_t* __restrict__ W,
                      int K, int mode,
                      bf16_t* __restrict__ phiQ, bf16_t* __restrict__ phiKT,
                      bf16_t* __restrict__ vT,
                      const float* __restrict__ bias,
                      bf16_t* __restrict__ gOut,
                      const float* __restrict__ xres, float* __restrict__ outp) {
    const int lane  = threadIdx.x & 31;
    const int wave  = threadIdx.x >> 5;
    const int waveM = wave >> 1;             // 0..3
    const int waveN = wave & 1;              // 0..1
    const long rowBase = (long)blockIdx.x * 128 + waveM * 32;
    const long colBase = (long)blockIdx.y * 128 + waveN * 64;
    const int ln16 = lane & 15;
    const int hl   = lane >> 4;

    Acc8 acc[2][4];
    #pragma unroll
    for (int mt = 0; mt < 2; ++mt)
        #pragma unroll
        for (int nt = 0; nt < 4; ++nt)
            #pragma unroll
            for (int e = 0; e < 8; ++e) acc[mt][nt].f[e] = 0.f;

    for (int k0 = 0; k0 < K; k0 += 32) {
        Frag16 a[2];
        #pragma unroll
        for (int mt = 0; mt < 2; ++mt) {
            const bf16_t* ap = A + (rowBase + mt * 16 + ln16) * (long)K + k0 + 8 * hl;
            a[mt].h[0] = *(const v8bf*)(ap);
            a[mt].h[1] = *(const v8bf*)(ap + 16);
        }
        Frag16 bf[4];
        #pragma unroll
        for (int nt = 0; nt < 4; ++nt) {
            const bf16_t* bp = W + (colBase + nt * 16 + ln16) * (long)K + k0 + 8 * hl;
            bf[nt].h[0] = *(const v8bf*)(bp);
            bf[nt].h[1] = *(const v8bf*)(bp + 16);
        }
        #pragma unroll
        for (int mt = 0; mt < 2; ++mt)
            #pragma unroll
            for (int nt = 0; nt < 4; ++nt)
                acc[mt][nt].v = __builtin_amdgcn_wmma_f32_16x16x32_bf16(
                    false, a[mt].v, false, bf[nt].v, (short)0, acc[mt][nt].v, false, false);
    }

    #pragma unroll
    for (int mt = 0; mt < 2; ++mt)
        #pragma unroll
        for (int nt = 0; nt < 4; ++nt)
            #pragma unroll
            for (int e = 0; e < 8; ++e) {
                const long row = rowBase + mt * 16 + e + 8 * hl;
                const long col = colBase + nt * 16 + ln16;
                const float av = acc[mt][nt].f[e];
                if (mode == 0) {                        // QKV epilogue
                    const long bi = row >> 12;
                    const long t  = row & 4095;
                    if (col < D_DIM) {
                        float p = (av >= 0.f) ? (av + 1.f) : __expf(av);   // elu+1
                        phiQ[row * D_DIM + col] = (bf16_t)p;
                    } else if (col < 2 * D_DIM) {
                        float p = (av >= 0.f) ? (av + 1.f) : __expf(av);
                        phiKT[(bi * D_DIM + (col - D_DIM)) * T_DIM + t] = (bf16_t)p;
                    } else {
                        vT[(bi * D_DIM + (col - 2 * D_DIM)) * T_DIM + t] = (bf16_t)av;
                    }
                } else if (mode == 1) {                 // gate epilogue
                    float s = 1.f / (1.f + __expf(-(av + bias[col])));
                    gOut[row * D_DIM + col] = (bf16_t)s;
                } else {                                 // out projection epilogue
                    outp[row * D_DIM + col] = xres[row * D_DIM + col] + av + bias[col];
                }
            }
}

// ---------------------------------------------------------------- kv = phiK^T @ V  (K split 8 ways)
__global__ __launch_bounds__(32)
void kv_partial_kernel(const bf16_t* __restrict__ phiKT, const bf16_t* __restrict__ vT,
                       float* __restrict__ kvPart) {
    const int bh    = blockIdx.x;          // b*16+h, 0..63
    const int split = blockIdx.y;          // 0..7
    const int b = bh >> 4, h = bh & 15;
    const int lane = threadIdx.x;
    const int ln16 = lane & 15, hl = lane >> 4;
    const bf16_t* Kb = phiKT + ((long)b * D_DIM + h * 64) * T_DIM;
    const bf16_t* Vb = vT    + ((long)b * D_DIM + h * 64) * T_DIM;
    const long t0b = (long)split * 512;

    Acc8 acc[4][4];
    #pragma unroll
    for (int mt = 0; mt < 4; ++mt)
        #pragma unroll
        for (int nt = 0; nt < 4; ++nt)
            #pragma unroll
            for (int e = 0; e < 8; ++e) acc[mt][nt].f[e] = 0.f;

    for (int k0 = 0; k0 < 512; k0 += 32) {
        const long t0 = t0b + k0;
        Frag16 a[4];
        #pragma unroll
        for (int mt = 0; mt < 4; ++mt) {
            const bf16_t* ap = Kb + (mt * 16 + ln16) * (long)T_DIM + t0 + 8 * hl;
            a[mt].h[0] = *(const v8bf*)(ap);
            a[mt].h[1] = *(const v8bf*)(ap + 16);
        }
        Frag16 bb[4];
        #pragma unroll
        for (int nt = 0; nt < 4; ++nt) {
            const bf16_t* bp = Vb + (nt * 16 + ln16) * (long)T_DIM + t0 + 8 * hl;
            bb[nt].h[0] = *(const v8bf*)(bp);
            bb[nt].h[1] = *(const v8bf*)(bp + 16);
        }
        #pragma unroll
        for (int mt = 0; mt < 4; ++mt)
            #pragma unroll
            for (int nt = 0; nt < 4; ++nt)
                acc[mt][nt].v = __builtin_amdgcn_wmma_f32_16x16x32_bf16(
                    false, a[mt].v, false, bb[nt].v, (short)0, acc[mt][nt].v, false, false);
    }

    float* dst = kvPart + ((long)bh * 8 + split) * 4096;
    #pragma unroll
    for (int mt = 0; mt < 4; ++mt)
        #pragma unroll
        for (int nt = 0; nt < 4; ++nt)
            #pragma unroll
            for (int e = 0; e < 8; ++e) {
                const int m = mt * 16 + e + 8 * hl;
                const int n = nt * 16 + ln16;
                dst[m * 64 + n] = acc[mt][nt].f[e];
            }
}

__global__ __launch_bounds__(256)
void kv_reduce_kernel(const float* __restrict__ kvPart, bf16_t* __restrict__ kvB) {
    const long i = (long)blockIdx.x * 256 + threadIdx.x;   // 64*4096
    if (i >= 64L * 4096) return;
    const long bh = i >> 12, r = i & 4095;
    float s = 0.f;
    #pragma unroll
    for (int p = 0; p < 8; ++p) s += kvPart[(bh * 8 + p) * 4096 + r];
    kvB[i] = (bf16_t)s;
}

// ---------------------------------------------------------------- depthwise conv (K=3, pad 1)
__global__ __launch_bounds__(256)
void conv_kernel(const bf16_t* __restrict__ xnB, const float* __restrict__ w,
                 const float* __restrict__ cb, bf16_t* __restrict__ yL) {
    const long idx = (long)blockIdx.x * 256 + threadIdx.x;   // 16M
    if (idx >= NROW * D_DIM) return;
    const long c = idx & (D_DIM - 1);
    const long row = idx >> 10;
    const long t = row & 4095;
    float a = cb[c];
    if (t > 0)    a += (float)xnB[(row - 1) * D_DIM + c] * w[c * 3 + 0];
    a += (float)xnB[row * D_DIM + c] * w[c * 3 + 1];
    if (t < 4095) a += (float)xnB[(row + 1) * D_DIM + c] * w[c * 3 + 2];
    yL[idx] = (bf16_t)a;
}

// ---------------------------------------------------------------- y_att = phiQ @ kv, fused gate mix
__global__ __launch_bounds__(256)
void yatt_mix_kernel(const bf16_t* __restrict__ phiQ, const bf16_t* __restrict__ kvB,
                     const bf16_t* __restrict__ gB, const bf16_t* __restrict__ yL,
                     bf16_t* __restrict__ yMix) {
    const int h = blockIdx.y;                    // 0..15
    const int lane = threadIdx.x & 31;
    const int wave = threadIdx.x >> 5;
    const long rowBase = (long)blockIdx.x * 256 + wave * 32;
    const long b = rowBase >> 12;
    const int ln16 = lane & 15, hl = lane >> 4;
    const bf16_t* kvh = kvB + ((long)b * 16 + h) * 4096;

    // kv fragments (register resident): 2 k-steps x 4 n-tiles
    Frag16 bb[2][4];
    #pragma unroll
    for (int ks = 0; ks < 2; ++ks)
        #pragma unroll
        for (int nt = 0; nt < 4; ++nt) {
            const int n = nt * 16 + ln16;
            #pragma unroll
            for (int j = 0; j < 16; ++j)
                bb[ks][nt].e[j] = kvh[(ks * 32 + 8 * hl + (j & 7) + 16 * (j >> 3)) * 64 + n];
        }

    Acc8 acc[2][4];
    #pragma unroll
    for (int mt = 0; mt < 2; ++mt)
        #pragma unroll
        for (int nt = 0; nt < 4; ++nt)
            #pragma unroll
            for (int e = 0; e < 8; ++e) acc[mt][nt].f[e] = 0.f;

    #pragma unroll
    for (int ks = 0; ks < 2; ++ks) {
        Frag16 a[2];
        #pragma unroll
        for (int mt = 0; mt < 2; ++mt) {
            const bf16_t* ap = phiQ + (rowBase + mt * 16 + ln16) * (long)D_DIM
                                    + h * 64 + ks * 32 + 8 * hl;
            a[mt].h[0] = *(const v8bf*)(ap);
            a[mt].h[1] = *(const v8bf*)(ap + 16);
        }
        #pragma unroll
        for (int mt = 0; mt < 2; ++mt)
            #pragma unroll
            for (int nt = 0; nt < 4; ++nt)
                acc[mt][nt].v = __builtin_amdgcn_wmma_f32_16x16x32_bf16(
                    false, a[mt].v, false, bb[ks][nt].v, (short)0, acc[mt][nt].v, false, false);
    }

    #pragma unroll
    for (int mt = 0; mt < 2; ++mt)
        #pragma unroll
        for (int nt = 0; nt < 4; ++nt)
            #pragma unroll
            for (int e = 0; e < 8; ++e) {
                const long row = rowBase + mt * 16 + e + 8 * hl;
                const long col = (long)h * 64 + nt * 16 + ln16;
                const float att = acc[mt][nt].f[e];
                const float g  = (float)gB[row * D_DIM + col];
                const float yl = (float)yL[row * D_DIM + col];
                yMix[row * D_DIM + col] = (bf16_t)(g * att + (1.f - g) * yl);
            }
}

// ---------------------------------------------------------------- launch
extern "C" void kernel_launch(void* const* d_in, const int* in_sizes, int n_in,
                              void* d_out, int out_size, void* d_ws, size_t ws_size,
                              hipStream_t stream) {
    const float* x      = (const float*)d_in[0];
    const float* ln_g   = (const float*)d_in[1];
    const float* ln_b   = (const float*)d_in[2];
    const float* w_qkv  = (const float*)d_in[3];
    const float* conv_w = (const float*)d_in[4];
    const float* conv_b = (const float*)d_in[5];
    const float* gln_g  = (const float*)d_in[6];
    const float* gln_b  = (const float*)d_in[7];
    const float* w_gate = (const float*)d_in[8];
    const float* b_gate = (const float*)d_in[9];
    const float* w_out  = (const float*)d_in[10];
    const float* b_out  = (const float*)d_in[11];
    float* outp = (float*)d_out;

    char* wsp = (char*)d_ws;
    auto take = [&](size_t bytes) -> char* {
        char* p = wsp; wsp += (bytes + 255) & ~(size_t)255; return p;
    };
    bf16_t* xnB    = (bf16_t*)take(NROW * D_DIM * 2);
    bf16_t* gxB    = (bf16_t*)take(NROW * D_DIM * 2);
    bf16_t* phiQ   = (bf16_t*)take(NROW * D_DIM * 2);
    bf16_t* phiKT  = (bf16_t*)take(NROW * D_DIM * 2);
    bf16_t* vT     = (bf16_t*)take(NROW * D_DIM * 2);
    bf16_t* gB     = (bf16_t*)take(NROW * D_DIM * 2);
    bf16_t* yL     = (bf16_t*)take(NROW * D_DIM * 2);
    bf16_t* yMix   = (bf16_t*)take(NROW * D_DIM * 2);
    bf16_t* wqkvB  = (bf16_t*)take(3072L * 1024 * 2);
    bf16_t* wgateB = (bf16_t*)take(1024L * 1024 * 2);
    bf16_t* woutB  = (bf16_t*)take(1024L * 1024 * 2);
    float*  kvPart = (float*) take(64L * 8 * 4096 * 4);
    bf16_t* kvB    = (bf16_t*)take(64L * 4096 * 2);

    // 1. weight conversion
    cvt_bf16_kernel<<<2048, 256, 0, stream>>>(w_qkv,  wqkvB,  3072L * 1024);
    cvt_bf16_kernel<<<1024, 256, 0, stream>>>(w_gate, wgateB, 1024L * 1024);
    cvt_bf16_kernel<<<1024, 256, 0, stream>>>(w_out,  woutB,  1024L * 1024);

    // 2. dual LayerNorm
    ln_dual_kernel<<<16384, 256, 0, stream>>>(x, ln_g, ln_b, gln_g, gln_b, xnB, gxB);

    // 3. QKV GEMM (N=3072) with phi + transposed K/V epilogue
    gemm_bf16_kernel<<<dim3(128, 24), 256, 0, stream>>>(
        xnB, wqkvB, 1024, 0, phiQ, phiKT, vT, nullptr, nullptr, nullptr, nullptr);

    // 4. gate GEMM (N=1024) with fused sigmoid
    gemm_bf16_kernel<<<dim3(128, 8), 256, 0, stream>>>(
        gxB, wgateB, 1024, 1, nullptr, nullptr, nullptr, b_gate, gB, nullptr, nullptr);

    // 5. kv state
    kv_partial_kernel<<<dim3(64, 8), 32, 0, stream>>>(phiKT, vT, kvPart);
    kv_reduce_kernel<<<1024, 256, 0, stream>>>(kvPart, kvB);

    // 6. depthwise conv
    conv_kernel<<<65536, 256, 0, stream>>>(xnB, conv_w, conv_b, yL);

    // 7. y_att + gate mix
    yatt_mix_kernel<<<dim3(64, 16), 256, 0, stream>>>(phiQ, kvB, gB, yL, yMix);

    // 8. output projection with residual
    gemm_bf16_kernel<<<dim3(128, 8), 256, 0, stream>>>(
        yMix, woutB, 1024, 2, nullptr, nullptr, nullptr, b_out, nullptr, x, outp);
}
